// QuantizedLinear_66108136620372
// MI455X (gfx1250) — matrix operand order repaired and checked by reading, not compiled, not run on hardware
//
#include <hip/hip_runtime.h>

#define IN_F 4096
#define OUT_F 4096
#define NUM_GROUPS 32
#define BM 128
#define BN 128
#define BK 64
#define NKC (IN_F / BK)   // 64 K-chunks
#define SA 72             // LDS row stride in halfs (144 B, 16B-aligned, bank-staggered)
#define SB 72

typedef __attribute__((ext_vector_type(16))) __bf16 v16bf;
typedef __attribute__((ext_vector_type(2)))  __bf16 v2bf;
typedef __attribute__((ext_vector_type(8)))  float  v8f;

union FragBF { uint4 q[2]; v16bf v; };

// fp32 pair -> packed bf16 (RNE). Single v_cvt_pk_bf16_f32 when available.
static __device__ __forceinline__ unsigned bfpair(float a, float b) {
#if __has_builtin(__builtin_amdgcn_cvt_pk_bf16_f32)
  v2bf v = __builtin_amdgcn_cvt_pk_bf16_f32(a, b);
  return __builtin_bit_cast(unsigned, v);
#else
  unsigned ua = __builtin_bit_cast(unsigned, a);
  unsigned ub = __builtin_bit_cast(unsigned, b);
  ua = (ua + 0x7FFFu + ((ua >> 16) & 1u)) >> 16;
  ub = (ub + 0x7FFFu + ((ub >> 16) & 1u));
  return (ua & 0x0000FFFFu) | (ub & 0xFFFF0000u);
#endif
}

__global__ __launch_bounds__(256)
void qlinear_bf16_wmma(const float* __restrict__ x,
                       const int*   __restrict__ wq,     // int8 values delivered as int32
                       const float* __restrict__ wscale, // [OUT_F][NUM_GROUPS]
                       const float* __restrict__ bias,   // [OUT_F]
                       float* __restrict__ out) {
  __shared__ unsigned short As[2][BM * SA];  // bf16 activations, [m][k]
  __shared__ unsigned short Bs[2][BN * SB];  // bf16 dequant weights, [n][k]

  const int tid    = threadIdx.x;
  const int wave   = tid >> 5;
  const int lane   = tid & 31;
  const int lane16 = lane & 15;
  const int laneHi = lane >> 4;
  const int wm     = wave & 3;   // 4 waves along M
  const int wn     = wave >> 2;  // 2 waves along N

  const int  blockN = blockIdx.x * BN;
  const long blockM = (long)blockIdx.y * BM;

  // A staging: each thread covers one float4-column (aCol4) of 8 rows (aRow + 16p)
  const int aCol4 = tid & 15;
  const int aRow  = tid >> 4;
  // B staging: each thread covers 16 consecutive int32 (bColI..+15) of 2 rows (bRow, bRow+64)
  const int bColI = (tid & 3) << 4;
  const int bRow  = tid >> 2;

  float4 ra[8];
  uint4  rb[2][4];
  float  sc[2];

  const float4* xg =
      reinterpret_cast<const float4*>(x) + (blockM + aRow) * (IN_F / 4) + aCol4;

  auto load_chunk = [&](int kc) {
    const float4* xp = xg + kc * (BK / 4);
#pragma unroll
    for (int p = 0; p < 8; ++p)
      ra[p] = xp[(size_t)p * 16 * (IN_F / 4)];
    const int g = kc >> 1;  // BK=64, GROUP=128
#pragma unroll
    for (int p = 0; p < 2; ++p) {
      const int ng = blockN + bRow + p * 64;
      const uint4* bp =
          reinterpret_cast<const uint4*>(wq + (size_t)ng * IN_F + kc * BK + bColI);
#pragma unroll
      for (int q = 0; q < 4; ++q) rb[p][q] = bp[q];
      sc[p] = wscale[ng * NUM_GROUPS + g];
    }
  };

  auto store_chunk = [&](int buf) {
#pragma unroll
    for (int p = 0; p < 8; ++p) {
      unsigned short* dp = &As[buf][(aRow + p * 16) * SA + aCol4 * 4];
      *reinterpret_cast<uint2*>(dp) =
          make_uint2(bfpair(ra[p].x, ra[p].y), bfpair(ra[p].z, ra[p].w));
    }
#pragma unroll
    for (int p = 0; p < 2; ++p) {
      const float s = sc[p];
      unsigned short* dp = &Bs[buf][(bRow + p * 64) * SB + bColI];
      unsigned w[8];
#pragma unroll
      for (int q = 0; q < 4; ++q) {
        const float f0 = (float)(int)rb[p][q].x * s;
        const float f1 = (float)(int)rb[p][q].y * s;
        const float f2 = (float)(int)rb[p][q].z * s;
        const float f3 = (float)(int)rb[p][q].w * s;
        w[2 * q]     = bfpair(f0, f1);
        w[2 * q + 1] = bfpair(f2, f3);
      }
      reinterpret_cast<uint4*>(dp)[0] = make_uint4(w[0], w[1], w[2], w[3]);
      reinterpret_cast<uint4*>(dp)[1] = make_uint4(w[4], w[5], w[6], w[7]);
    }
  };

  v8f acc[2][4];
  {
    v8f z = {};
#pragma unroll
    for (int t = 0; t < 2; ++t)
#pragma unroll
      for (int j = 0; j < 4; ++j) acc[t][j] = z;
  }

  auto compute = [&](int buf) {
#pragma unroll
    for (int ks = 0; ks < 2; ++ks) {
      const int kb = ks * 32;
      FragBF a[2], b[4];
      // A frag (16x32 bf16): lane = M%16; halfs 0..7 -> K = kb + laneHi*8 + 0..7,
      // halfs 8..15 -> K = kb + 16 + laneHi*8 + 0..7  => two ds_load_b128
#pragma unroll
      for (int t = 0; t < 2; ++t) {
        const unsigned short* p =
            &As[buf][(wm * 32 + t * 16 + lane16) * SA + kb + laneHi * 8];
        a[t].q[0] = *reinterpret_cast<const uint4*>(p);
        a[t].q[1] = *reinterpret_cast<const uint4*>(p + 16);
      }
      // B frag (32x16 bf16): lane = N%16; halfs 0..7 -> K = kb + laneHi*8 + 0..7, halfs 8..15 -> K = kb + 16 + laneHi*8 + 0..7
#pragma unroll
      for (int j = 0; j < 4; ++j) {
        const unsigned short* p =
            &Bs[buf][(wn * 64 + j * 16 + lane16) * SB + kb + laneHi * 8];
        b[j].q[0] = *reinterpret_cast<const uint4*>(p);
        b[j].q[1] = *reinterpret_cast<const uint4*>(p + 16);
      }
#pragma unroll
      for (int t = 0; t < 2; ++t)
#pragma unroll
        for (int j = 0; j < 4; ++j)
          acc[t][j] = __builtin_amdgcn_wmma_f32_16x16x32_bf16(
              false, a[t].v, false, b[j].v, (short)0, acc[t][j], false, false);
    }
  };

  // Software pipeline, single barrier per chunk:
  //   [global loads k+1 in flight] -> WMMA on buf -> convert+store k+1 into buf^1 -> barrier
  load_chunk(0);
  store_chunk(0);
  __syncthreads();
  int buf = 0;
  for (int kc = 0; kc < NKC; ++kc) {
    const bool more = (kc + 1 < NKC);
    if (more) load_chunk(kc + 1);   // HBM latency hidden under the WMMA block below
    compute(buf);
    if (more) store_chunk(buf ^ 1); // loadcnt wait lands here, after all WMMAs
    __syncthreads();
    buf ^= 1;
  }

  // Epilogue: C/D layout is VGPR r -> M = r (lanes 0-15) / r+8 (lanes 16-31), N = lane%16.
  // Output has no reuse: store non-temporally to keep x/w resident in L2.
#pragma unroll
  for (int t = 0; t < 2; ++t) {
    const long m0 = blockM + wm * 32 + t * 16 + laneHi * 8;
#pragma unroll
    for (int j = 0; j < 4; ++j) {
      const int   n  = blockN + wn * 64 + j * 16 + lane16;
      const float bv = bias[n];
      float* op = out + m0 * OUT_F + n;
#pragma unroll
      for (int r = 0; r < 8; ++r)
        __builtin_nontemporal_store(acc[t][j][r] + bv, op + (long)r * OUT_F);
    }
  }
}

extern "C" void kernel_launch(void* const* d_in, const int* in_sizes, int n_in,
                              void* d_out, int out_size, void* d_ws, size_t ws_size,
                              hipStream_t stream) {
  const float* x      = (const float*)d_in[0];
  const int*   wq     = (const int*)d_in[1];   // integer input -> const int* per harness contract
  const float* wscale = (const float*)d_in[2];
  const float* bias   = (const float*)d_in[3];
  float*       out    = (float*)d_out;

  const int Mtotal = in_sizes[0] / IN_F;  // 4*2048 = 8192
  dim3 grid(OUT_F / BN, Mtotal / BM);     // (32, 64)
  qlinear_bf16_wmma<<<grid, dim3(256), 0, stream>>>(x, wq, wscale, bias, out);
}
